// KroneckerAttention_7078106104253
// MI455X (gfx1250) — matrix operand order repaired and checked by reading, not compiled, not run on hardware
//
#include <hip/hip_runtime.h>
#include <math.h>

// Problem constants (from reference): B=2,H=16,NQ=NK=4096,D=64,m=n=4,pq=pk=1024,c0=c1=2
#define NQK    4096
#define DIM    64
#define PQ     1024
#define NGRP   4

typedef __bf16 bf16_t;
typedef __attribute__((ext_vector_type(16))) __bf16 v16bf;
typedef __attribute__((ext_vector_type(8)))  float  v8f;

union Frag16 { v16bf v; unsigned int u[8]; };

__device__ __forceinline__ unsigned short f2bf(float f) {
    unsigned int u = __float_as_uint(f);
    u += 0x7FFFu + ((u >> 16) & 1u);   // round-to-nearest-even
    return (unsigned short)(u >> 16);
}
__device__ __forceinline__ unsigned int pack2bf(float lo, float hi) {
    return ((unsigned int)f2bf(hi) << 16) | (unsigned int)f2bf(lo);
}

// Raw v_exp_f32 (valid here: |S| <= ~8, no range fixup needed)
__device__ __forceinline__ float fast_exp2(float x) {
    return __builtin_amdgcn_exp2f(x);
}

__device__ __forceinline__ v8f wmma_bf16(v16bf a, v16bf b, v8f c) {
    // D(16x16,f32) = A(16x32,bf16) * B(32x16,bf16) + C
    return __builtin_amdgcn_wmma_f32_16x16x32_bf16(
        false, a, false, b, (short)0, c, false, false);
}

// A-matrix (16x32 bf16) fragment from row-major LDS tile (ISA 7.12.2 layout).
__device__ __forceinline__ v16bf load_a_frag(const unsigned short* p, int lane, int pitch) {
    int row = lane & 15;
    int hk  = (lane >> 4) << 3;
    Frag16 f;
    const unsigned short* r = p + row * pitch;
#pragma unroll
    for (int i = 0; i < 4; ++i) {
        f.u[i]     = *(const unsigned int*)(r + hk + 2 * i);
        f.u[4 + i] = *(const unsigned int*)(r + 16 + hk + 2 * i);
    }
    return f.v;
}

// B-matrix (32x16 bf16) fragment; LDS holds B^T row-major (row = B column N,
// contiguous along the contraction dim K).
__device__ __forceinline__ v16bf load_b_frag(const unsigned short* p, int lane, int pitch) {
    int col = lane & 15;
    int kst = (lane >> 4) << 3;
    Frag16 f;
    const unsigned short* r = p + col * pitch + kst;
#pragma unroll
    for (int i = 0; i < 8; ++i) f.u[i] = *(const unsigned int*)(r + 2 * (i & 3) + ((i >> 2) << 4));
    return f.v;
}

// ---------------- Phase A1: v_sum[bh][n][d] ----------------
__global__ void vsum_kernel(const float* __restrict__ value, float* __restrict__ vsum) {
    int idx = blockIdx.x * 256 + threadIdx.x;        // bh*256 + n*64 + d
    int bh = idx >> 8, n = (idx >> 6) & 3, d = idx & 63;
    const float* base = value + ((size_t)bh * NQK + n * PQ) * DIM + d;
    float s = 0.f;
    for (int k = 0; k < PQ; ++k) s += base[(size_t)k * DIM];
    vsum[idx] = s;
}

// ---------------- Phase A2: norms[bh][m][j] ----------------
__global__ void norm_kernel(const float* __restrict__ query, const float* __restrict__ key,
                            float* __restrict__ norms) {
    __shared__ float ks[NGRP][DIM];
    __shared__ float red[256];
    int bh = blockIdx.x >> 2, m = blockIdx.x & 3;
    int tid = threadIdx.x;
    { int j = tid >> 6, d = tid & 63;
      ks[j][d] = key[((size_t)bh * NQK + j * PQ) * DIM + d]; }
    __syncthreads();
    float sq[4] = {0.f, 0.f, 0.f, 0.f};
    for (int qi = 0; qi < 4; ++qi) {
        int q = m * PQ + qi * 256 + tid;
        const float* qrow = query + ((size_t)bh * NQK + q) * DIM;
        float w[4] = {0.f, 0.f, 0.f, 0.f};
        for (int d = 0; d < DIM; ++d) {
            float qv = qrow[d];
#pragma unroll
            for (int j = 0; j < 4; ++j) w[j] += qv * ks[j][d];
        }
#pragma unroll
        for (int j = 0; j < 4; ++j) sq[j] += w[j] * w[j];
    }
    for (int j = 0; j < 4; ++j) {
        red[tid] = sq[j]; __syncthreads();
        for (int s = 128; s > 0; s >>= 1) {
            if (tid < s) red[tid] += red[tid + s];
            __syncthreads();
        }
        if (tid == 0) norms[bh * 16 + m * 4 + j] = sqrtf(red[0]);
        __syncthreads();
    }
}

// ---------------- Phase B: pipelined WMMA attention ----------------
// Contraction-dim permutation: k-within-chunk stored at kpos = 2*(k&15) + (k>>4).
// Applied identically to P columns (writer) and V rows (stager), so P.V is unchanged.
struct ChunkRegs { float4 k[2]; float4 va[4]; float4 vb[4]; };

__device__ __forceinline__ void chunk_load(ChunkRegs& c, const float* __restrict__ key,
                                           const float* __restrict__ value,
                                           size_t kbase, size_t vbase, int kc, int tid) {
#pragma unroll
    for (int i = 0; i < 2; ++i) {
        int e4 = i * 256 + tid;                  // 512 float4 = 32k x 64d
        int k = e4 >> 4, d = (e4 & 15) << 2;
        c.k[i] = *(const float4*)(key + kbase + (size_t)(kc * 32 + k) * DIM + d);
    }
#pragma unroll
    for (int i = 0; i < 4; ++i) {
        int pe = i * 256 + tid;                  // 1024 pair-quads = 4n x 16k' x 16d4
        int n = pe >> 8, kp = (pe >> 4) & 15, d = (pe & 15) << 2;
        const float* vp = value + vbase + (size_t)(n * PQ + kc * 32 + kp) * DIM + d;
        c.va[i] = *(const float4*)(vp);
        c.vb[i] = *(const float4*)(vp + 16 * DIM);   // k' + 16
    }
}

__device__ __forceinline__ void chunk_store(const ChunkRegs& c, unsigned short* Kb,
                                            unsigned short* Vb, int tid) {
#pragma unroll
    for (int i = 0; i < 2; ++i) {
        int e4 = i * 256 + tid;
        int k = e4 >> 4, d = (e4 & 15) << 2;
        uint2 u;
        u.x = pack2bf(c.k[i].x, c.k[i].y);
        u.y = pack2bf(c.k[i].z, c.k[i].w);
        *(uint2*)(Kb + k * DIM + d) = u;         // 8B aligned (d % 4 == 0)
    }
#pragma unroll
    for (int i = 0; i < 4; ++i) {
        int pe = i * 256 + tid;
        int n = pe >> 8, kp = (pe >> 4) & 15, d = (pe & 15) << 2;
        float a[4] = {c.va[i].x, c.va[i].y, c.va[i].z, c.va[i].w};
        float b[4] = {c.vb[i].x, c.vb[i].y, c.vb[i].z, c.vb[i].w};
#pragma unroll
        for (int j = 0; j < 4; ++j)              // kpos(k')=2k', kpos(k'+16)=2k'+1 adjacent
            *(unsigned int*)(Vb + (n * DIM + d + j) * 32 + 2 * kp) = pack2bf(a[j], b[j]);
    }
}

__device__ __forceinline__ void chunk_prefetch(const float* __restrict__ key,
                                               const float* __restrict__ value,
                                               size_t kbase, size_t vbase, int kc, int tid) {
    {
        int k = tid >> 4, d = (tid & 15) << 2;
        __builtin_prefetch(key + kbase + (size_t)(kc * 32 + k) * DIM + d, 0, 3);
    }
#pragma unroll
    for (int i = 0; i < 4; ++i) {
        int pe = i * 256 + tid;
        int n = pe >> 8, kp = (pe >> 4) & 15, d = (pe & 15) << 2;
        const float* vp = value + vbase + (size_t)(n * PQ + kc * 32 + kp) * DIM + d;
        __builtin_prefetch(vp, 0, 3);
        __builtin_prefetch(vp + 16 * DIM, 0, 3);
    }
}

__global__ __launch_bounds__(256, 1)
void attn_kernel(const float* __restrict__ query, const float* __restrict__ key,
                 const float* __restrict__ value, float* __restrict__ unnorm,
                 float* __restrict__ den) {
    // 48 KB shared, regions reused across phases:
    //   Q phase : [0, 8192)  = 128x64 bf16 Q tile
    //   main    : Ks0 [0,2048) Ks1 [2048,4096) Vs0 [4096,12288) Vs1 [12288,20480) Ps [20480,24576)
    __shared__ unsigned short smem[24576];
    unsigned short* Qs  = smem;
    unsigned short* Ks0 = smem;
    unsigned short* Ks1 = smem + 2048;
    unsigned short* Vs0 = smem + 4096;
    unsigned short* Vs1 = smem + 12288;
    unsigned short* Ps  = smem + 20480;

    const int bh = blockIdx.x >> 3;
    const int qtile = blockIdx.x & 7;
    const int tid = threadIdx.x;
    const int wave = tid >> 5;
    const int lane = tid & 31;
    // fold softmax scale (D^-0.5 = 1/8) and log2(e) into Q so P = exp2(S)
    const float QSCALE = 0.125f * 1.4426950408889634f;

    const size_t qoff  = ((size_t)bh * NQK + 2 * PQ + qtile * 128) * DIM;
    const size_t kbase = ((size_t)bh * NQK + 2 * PQ) * DIM;
    const size_t vbase = (size_t)bh * NQK * DIM;

    // Prefetch chunk 0 into registers while staging Q.
    ChunkRegs cr;
    chunk_load(cr, key, value, kbase, vbase, 0, tid);

#pragma unroll
    for (int i = 0; i < 8; ++i) {                // 2048 float4 of Q
        int e4 = i * 256 + tid;
        int r = e4 >> 4, d = (e4 & 15) << 2;
        float4 q = *(const float4*)(query + qoff + (size_t)r * DIM + d);
        uint2 u;
        u.x = pack2bf(q.x * QSCALE, q.y * QSCALE);
        u.y = pack2bf(q.z * QSCALE, q.w * QSCALE);
        *(uint2*)(Qs + r * DIM + d) = u;
    }
    __syncthreads();
    v16bf aq0 = load_a_frag(Qs + (wave * 16) * DIM + 0,  lane, DIM);
    v16bf aq1 = load_a_frag(Qs + (wave * 16) * DIM + 32, lane, DIM);
    __syncthreads();                             // Qs dead; region becomes K/V buffers
    chunk_store(cr, Ks0, Vs0, tid);
    __syncthreads();

    Frag16 onesf;
#pragma unroll
    for (int i = 0; i < 8; ++i) onesf.u[i] = 0x3F803F80u;   // packed bf16 1.0
    const v16bf bones = onesf.v;

    const v8f vzero = {0.f, 0.f, 0.f, 0.f, 0.f, 0.f, 0.f, 0.f};
    v8f acc[16];
#pragma unroll
    for (int t = 0; t < 16; ++t) acc[t] = vzero;
    v8f dacc = vzero;

    for (int kc = 0; kc < 32; ++kc) {
        unsigned short* Kc = (kc & 1) ? Ks1 : Ks0;
        unsigned short* Vc = (kc & 1) ? Vs1 : Vs0;
        const bool more = (kc + 1) < 32;
        if (more) chunk_load(cr, key, value, kbase, vbase, kc + 1, tid);  // in flight
        if (kc + 2 < 32) chunk_prefetch(key, value, kbase, vbase, kc + 2, tid);

        // S(16x32) = Qs . K^T (pre-scaled)
        v8f s0 = vzero, s1 = vzero;
        {
            v16bf b00 = load_b_frag(Kc + 0,             lane, DIM);
            v16bf b01 = load_b_frag(Kc + 32,            lane, DIM);
            v16bf b10 = load_b_frag(Kc + 16 * DIM,      lane, DIM);
            v16bf b11 = load_b_frag(Kc + 16 * DIM + 32, lane, DIM);
            s0 = wmma_bf16(aq0, b00, s0);
            s0 = wmma_bf16(aq1, b01, s0);
            s1 = wmma_bf16(aq0, b10, s1);
            s1 = wmma_bf16(aq1, b11, s1);
        }

        // P = exp2(S) via bare v_exp_f32; packed pair store (cols N,N+16 -> kpos 2N,2N+1)
        unsigned short* pp = Ps + wave * 512;
#pragma unroll
        for (int r = 0; r < 8; ++r) {
            int M = r + ((lane >> 4) << 3);
            int N = lane & 15;
            *(unsigned int*)(pp + M * 32 + 2 * N) =
                pack2bf(fast_exp2(s0[r]), fast_exp2(s1[r]));
        }
        v16bf pa = load_a_frag(pp, lane, 32);

        dacc = wmma_bf16(pa, bones, dacc);       // row sums -> den
#pragma unroll
        for (int t = 0; t < 16; ++t) {           // unnorm(16x256) += P . V_cat
            v16bf bv = load_b_frag(Vc + (t * 16) * 32, lane, 32);
            acc[t] = wmma_bf16(pa, bv, acc[t]);
        }

        if (more) chunk_store(cr, (kc & 1) ? Ks0 : Ks1, (kc & 1) ? Vs0 : Vs1, tid);
        __syncthreads();
    }

    // Epilogue: C layout (M = r + 8*(lane>=16), N = lane&15) -> global
    const int M0 = (lane >> 4) << 3;
    const int N  = lane & 15;
    const size_t qrowbase = (size_t)bh * PQ + qtile * 128 + wave * 16;
#pragma unroll
    for (int t = 0; t < 16; ++t)
#pragma unroll
        for (int r = 0; r < 8; ++r)
            unnorm[(qrowbase + (size_t)(r + M0)) * 256 + t * 16 + N] = acc[t][r];
    if (N == 0) {
#pragma unroll
        for (int r = 0; r < 8; ++r) den[qrowbase + r + M0] = dacc[r];
    }
}

// ---------------- Phase C: combine ----------------
__global__ void combine_kernel(const float* __restrict__ unnorm, const float* __restrict__ den,
                               const float* __restrict__ vsum, const float* __restrict__ norms,
                               float* __restrict__ out) {
    int tid = threadIdx.x;
    int r = tid >> 6, d = tid & 63;
    int row = blockIdx.x * 4 + r;            // bh*4096 + m*1024 + q
    int bh = row >> 12;
    int qq = row & 4095;
    int m = qq >> 10;
    int q = qq & 1023;

    const float* nb = norms + bh * 16;
    float ref = nb[2 * 4 + 2];
    float sg[4], om[4];
    float dden = 0.f;
#pragma unroll
    for (int j = 0; j < 4; ++j) {
        float s = nb[m * 4 + j] / ref;
        sg[j] = s;
        om[j] = fmaxf(1.0f - s, 0.0f);
        dden += om[j] * 1024.0f;             // om * pk
    }
    float dq = den[bh * 1024 + q];
    float num = 0.f;
#pragma unroll
    for (int j = 0; j < 4; ++j) {
        num  += om[j] * vsum[bh * 256 + j * 64 + d]
              + sg[j] * unnorm[((size_t)(bh * 1024 + q)) * 256 + j * 64 + d];
        dden += sg[j] * dq;
    }
    out[(size_t)row * 64 + d] = num / dden;
    if (d == 0) out[8388608 + row] = __logf(dden);   // lse
}

extern "C" void kernel_launch(void* const* d_in, const int* in_sizes, int n_in,
                              void* d_out, int out_size, void* d_ws, size_t ws_size,
                              hipStream_t stream) {
    const float* query = (const float*)d_in[0];
    const float* key   = (const float*)d_in[1];
    const float* value = (const float*)d_in[2];
    float* out = (float*)d_out;

    float* ws     = (float*)d_ws;
    float* unnorm = ws;                      // 8,388,608 floats
    float* den    = unnorm + 8388608;        //    32,768 floats
    float* vsum   = den + 32768;             //     8,192 floats
    float* norms  = vsum + 8192;             //       512 floats

    hipLaunchKernelGGL(vsum_kernel,    dim3(32),    dim3(256), 0, stream, value, vsum);
    hipLaunchKernelGGL(norm_kernel,    dim3(128),   dim3(256), 0, stream, query, key, norms);
    hipLaunchKernelGGL(attn_kernel,    dim3(256),   dim3(256), 0, stream,
                       query, key, value, unnorm, den);
    hipLaunchKernelGGL(combine_kernel, dim3(32768), dim3(256), 0, stream,
                       unnorm, den, vsum, norms, out);
}
